// BlockGRUCell_24163486008054
// MI455X (gfx1250) — matrix operand order repaired and checked by reading, not compiled, not run on hardware
//
#include <hip/hip_runtime.h>

// ---------------------------------------------------------------------------
// BlockGRUCell on gfx1250 (MI455X): double-buffered WMMA f16 GEMMs + fp32 epilogues.
// B=2048, D=4096, S=1024, A=128, H=1024, G=8, DG=512, IN_G=3584
// ---------------------------------------------------------------------------

#define BM 128
#define BN 128
#define BK 32
#define LDS_STRIDE 40   // halves; 80B row stride keeps 16B alignment for b128 LDS loads

typedef __attribute__((ext_vector_type(16))) _Float16 v16h;
typedef __attribute__((ext_vector_type(8)))  _Float16 v8h;
typedef __attribute__((ext_vector_type(4)))  _Float16 v4h;
typedef __attribute__((ext_vector_type(8)))  float    v8f;
typedef __attribute__((ext_vector_type(4)))  float    v4f;

union V16 { v16h v; v8h h[2]; };

// Stage next K-step's tiles from global into registers (raw fp32 so the
// loadcnt waits can sink past the WMMA block).
__device__ __forceinline__ void stage_global(const float* __restrict__ Ap, int la, int kcol, int m0,
                                             const float* __restrict__ Bw, int ldb, int kb, int n0,
                                             int t, v4f* sa, v4f* sb)
{
#pragma unroll
    for (int i = 0; i < 4; ++i) {
        int fid = t + (i << 8);            // 0..1023 float4 slots of 128x32 A tile
        int row = fid >> 3;
        int col = (fid & 7) << 2;
        const float* p = Ap + (size_t)(m0 + row) * la + kcol + col;
        __builtin_prefetch(p + BK, 0, 3);
        sa[i] = *(const v4f*)p;
    }
    // B tile 32x128: thread owns 4 consecutive K rows at one N column-quad
    int kq  = t >> 5;                      // 0..7
    int col = (t & 31) << 2;               // 0..124
#pragma unroll
    for (int i = 0; i < 4; ++i) {
        const float* p = Bw + (size_t)(kb + kq * 4 + i) * ldb + n0 + col;
        __builtin_prefetch(p + (size_t)BK * ldb, 0, 3);
        sb[i] = *(const v4f*)p;
    }
}

// Convert staged registers to f16 and store into one LDS buffer.
// A: row-major [row][k]. B: transposed [n][k] via in-register 4x4 transpose,
// so stores are ds_store_b64 (no scalar b16 stores).
__device__ __forceinline__ void stage_to_lds(_Float16* __restrict__ As, _Float16* __restrict__ Bs,
                                             int t, const v4f* sa, const v4f* sb)
{
#pragma unroll
    for (int i = 0; i < 4; ++i) {
        int fid = t + (i << 8);
        int row = fid >> 3;
        int col = (fid & 7) << 2;
        v4h h = { (_Float16)sa[i][0], (_Float16)sa[i][1], (_Float16)sa[i][2], (_Float16)sa[i][3] };
        *(v4h*)&As[row * LDS_STRIDE + col] = h;
    }
    int kq  = t >> 5;
    int col = (t & 31) << 2;
#pragma unroll
    for (int j = 0; j < 4; ++j) {
        v4h h = { (_Float16)sb[0][j], (_Float16)sb[1][j], (_Float16)sb[2][j], (_Float16)sb[3][j] };
        *(v4h*)&Bs[(col + j) * LDS_STRIDE + kq * 4] = h;   // 8B column store
    }
}

// Generic WMMA GEMM: C[M,N] = A[M,K] * B[K,N], A virtually split:
//   cols [0,K0) from A0 (stride lda0), cols [K0,K) from A1 (stride lda1).
// blockIdx.z = group with per-group element offsets a0g / a1g / bg / cg.
__launch_bounds__(256)
__global__ void wmma_gemm_split(const float* __restrict__ A0, int lda0, int K0, long long a0g,
                                const float* __restrict__ A1, int lda1, long long a1g,
                                const float* __restrict__ Bw, int ldb, long long bg,
                                float* __restrict__ C, int ldc, long long cg,
                                int K)
{
    const int g = blockIdx.z;
    A0 += (long long)g * a0g;
    A1 += (long long)g * a1g;
    Bw += (long long)g * bg;
    C  += (long long)g * cg;

    const int m0 = blockIdx.y * BM;
    const int n0 = blockIdx.x * BN;

    __shared__ _Float16 As[2][BM * LDS_STRIDE];   // [buf][row][k] 128x32(+pad)
    __shared__ _Float16 Bs[2][BN * LDS_STRIDE];   // [buf][col][k] transposed

    const int t     = threadIdx.x;     // 0..255  (8 waves of 32)
    const int wave  = t >> 5;
    const int lane  = t & 31;
    const int wm    = wave >> 2;       // 0..1 -> 64-row strip
    const int wn    = wave & 3;        // 0..3 -> 32-col strip
    const int l15   = lane & 15;
    const int khalf = lane >> 4;       // 0 or 1

    v8f acc[4][2] = {};                // 4 M-tiles x 2 N-tiles of 16x16 f32

    v4f sa[4], sb[4];
    // prologue: stage + fill buffer 0
    stage_global(A0, lda0, 0, m0, Bw, ldb, 0, n0, t, sa, sb);
    stage_to_lds(As[0], Bs[0], t, sa, sb);

    int cur = 0;
    for (int kb = 0; kb < K; kb += BK) {
        const int nxt = kb + BK;
        const bool have_next = nxt < K;
        if (have_next) {
            const float* Ap; int la; int kcol;
            if (nxt < K0) { Ap = A0; la = lda0; kcol = nxt;      }
            else          { Ap = A1; la = lda1; kcol = nxt - K0; }
            stage_global(Ap, la, kcol, m0, Bw, ldb, nxt, n0, t, sa, sb);
        }

        __syncthreads();               // buf[cur] writes visible; buf[cur^1] free

        const _Float16* Ab = As[cur];
        const _Float16* Bb = Bs[cur];

        // ---- fragments per ISA 7.12.2 layouts ----
        V16 af[4], bf[2];
#pragma unroll
        for (int mt = 0; mt < 4; ++mt) {
            int r = wm * 64 + mt * 16 + l15;
            // A 16x32: lane half 0 -> K 0..7 & 16..23 ; half 1 -> K 8..15 & 24..31
            af[mt].h[0] = *(const v8h*)&Ab[r * LDS_STRIDE + khalf * 8];
            af[mt].h[1] = *(const v8h*)&Ab[r * LDS_STRIDE + khalf * 8 + 16];
        }
#pragma unroll
        for (int nt = 0; nt < 2; ++nt) {
            int c = wn * 32 + nt * 16 + l15;
            // B 32x16: lane half 0 -> K 0..7 & 16..23 ; half 1 -> K 8..15 & 24..31
            bf[nt].h[0] = *(const v8h*)&Bb[c * LDS_STRIDE + khalf * 8];
            bf[nt].h[1] = *(const v8h*)&Bb[c * LDS_STRIDE + khalf * 8 + 16];
        }

#pragma unroll
        for (int mt = 0; mt < 4; ++mt)
#pragma unroll
            for (int nt = 0; nt < 2; ++nt)
                acc[mt][nt] = __builtin_amdgcn_wmma_f32_16x16x32_f16(
                    false, af[mt].v, false, bf[nt].v,
                    (short)0, acc[mt][nt], false, false);

        if (have_next) stage_to_lds(As[cur ^ 1], Bs[cur ^ 1], t, sa, sb);
        cur ^= 1;
    }

    // ---- store C (fp32). VGPR e: row = tile + khalf*8 + e, col = tile + l15 ----
    const int rbase = m0 + wm * 64 + khalf * 8;
    const int cbase = n0 + wn * 32 + l15;
#pragma unroll
    for (int mt = 0; mt < 4; ++mt)
#pragma unroll
        for (int nt = 0; nt < 2; ++nt)
#pragma unroll
            for (int e = 0; e < 8; ++e)
                C[(size_t)(rbase + mt * 16 + e) * ldc + cbase + nt * 16] = acc[mt][nt][e];
}

// ---------------------------------------------------------------------------
// In-place bias + LayerNorm + SiLU over a width-W row chunk. One block per row.
// ---------------------------------------------------------------------------
__launch_bounds__(256)
__global__ void ln_silu_kernel(float* __restrict__ X, int ldx,
                               const float* __restrict__ bias,
                               const float* __restrict__ gamma,
                               const float* __restrict__ beta,
                               int W)
{
    const int row = blockIdx.x;
    float* x = X + (size_t)row * ldx;
    const int t = threadIdx.x;
    const int nper = W >> 8;          // 4 (W=1024) or 16 (W=4096)

    __shared__ float red[256];

    float vals[16];
    float s = 0.f;
    for (int i = 0; i < nper; ++i) {
        int c = t + (i << 8);
        float v = x[c] + bias[c];
        vals[i] = v;
        s += v;
    }
    red[t] = s; __syncthreads();
    for (int off = 128; off > 0; off >>= 1) {
        if (t < off) red[t] += red[t + off];
        __syncthreads();
    }
    const float mean = red[0] / (float)W;
    __syncthreads();

    float vs = 0.f;
    for (int i = 0; i < nper; ++i) { float d = vals[i] - mean; vs += d * d; }
    red[t] = vs; __syncthreads();
    for (int off = 128; off > 0; off >>= 1) {
        if (t < off) red[t] += red[t + off];
        __syncthreads();
    }
    const float rstd = rsqrtf(red[0] / (float)W + 1e-5f);

    for (int i = 0; i < nper; ++i) {
        int c = t + (i << 8);
        float v = (vals[i] - mean) * rstd * gamma[c] + beta[c];
        v = v / (1.f + __expf(-v));   // SiLU
        x[c] = v;
    }
}

// ---------------------------------------------------------------------------
// GRU gating: out = sigmoid(u-1)*tanh(sigmoid(r)*c) + (1-sigmoid(u-1))*deter
// ---------------------------------------------------------------------------
__launch_bounds__(256)
__global__ void gru_gate_kernel(const float* __restrict__ z,
                                const float* __restrict__ gru_b,
                                const float* __restrict__ deter,
                                float* __restrict__ out)
{
    const int i = blockIdx.x * 256 + threadIdx.x;     // over B*D
    const int d = i & 4095;
    const size_t zrow = (size_t)(i >> 12) * 12288;

    float r = z[zrow + d]        + gru_b[d];
    float c = z[zrow + 4096 + d] + gru_b[4096 + d];
    float u = z[zrow + 8192 + d] + gru_b[8192 + d] - 1.0f;

    r = 1.f / (1.f + __expf(-r));
    u = 1.f / (1.f + __expf(-u));
    c = tanhf(r * c);

    out[i] = u * c + (1.f - u) * deter[i];
}

// ---------------------------------------------------------------------------
extern "C" void kernel_launch(void* const* d_in, const int* in_sizes, int n_in,
                              void* d_out, int out_size, void* d_ws, size_t ws_size,
                              hipStream_t stream)
{
    const int B = 2048, D = 4096, S = 1024, A = 128, H = 1024, G = 8;
    const int DG = D / G;            // 512
    const int ING = DG + 3 * H;      // 3584
    const int H3 = 3 * H;            // 3072
    const int D3 = 3 * D;            // 12288

    const float* deter  = (const float*)d_in[0];
    const float* stoch  = (const float*)d_in[1];
    const float* action = (const float*)d_in[2];
    const float* W_d  = (const float*)d_in[3];
    const float* b_d  = (const float*)d_in[4];
    const float* g_d  = (const float*)d_in[5];
    const float* be_d = (const float*)d_in[6];
    const float* W_s  = (const float*)d_in[7];
    const float* b_s  = (const float*)d_in[8];
    const float* g_s  = (const float*)d_in[9];
    const float* be_s = (const float*)d_in[10];
    const float* W_a  = (const float*)d_in[11];
    const float* b_a  = (const float*)d_in[12];
    const float* g_a  = (const float*)d_in[13];
    const float* be_a = (const float*)d_in[14];
    const float* dyn_W  = (const float*)d_in[15];
    const float* dyn_b  = (const float*)d_in[16];
    const float* g_dyn  = (const float*)d_in[17];
    const float* be_dyn = (const float*)d_in[18];
    const float* gru_W  = (const float*)d_in[19];
    const float* gru_b  = (const float*)d_in[20];
    float* out = (float*)d_out;

    // workspace layout (fp32): xpre [B,3H] | ypre [B,D] | z [B,3D]
    float* xpre = (float*)d_ws;
    float* ypre = xpre + (size_t)B * H3;
    float* z    = ypre + (size_t)B * D;

    const dim3 blk(256);

    // ---- input projections -> xpre columns [0,H) [H,2H) [2H,3H) ----
    {
        dim3 grid(H / BN, B / BM, 1);
        wmma_gemm_split<<<grid, blk, 0, stream>>>(deter,  D, D, 0, deter,  D, 0,
                                                  W_d, H, 0, xpre,          H3, 0, D);
        wmma_gemm_split<<<grid, blk, 0, stream>>>(stoch,  S, S, 0, stoch,  S, 0,
                                                  W_s, H, 0, xpre + H,      H3, 0, S);
        wmma_gemm_split<<<grid, blk, 0, stream>>>(action, A, A, 0, action, A, 0,
                                                  W_a, H, 0, xpre + 2 * H,  H3, 0, A);
    }

    // ---- LN + SiLU on the three H-wide chunks (in place) ----
    ln_silu_kernel<<<B, blk, 0, stream>>>(xpre,         H3, b_d, g_d, be_d, H);
    ln_silu_kernel<<<B, blk, 0, stream>>>(xpre + H,     H3, b_s, g_s, be_s, H);
    ln_silu_kernel<<<B, blk, 0, stream>>>(xpre + 2 * H, H3, b_a, g_a, be_a, H);

    // ---- dyn layer: per group, A = [deter[:,g*DG:(g+1)*DG] | x], B = dyn_W[g] ----
    {
        dim3 grid(DG / BN, B / BM, G);
        wmma_gemm_split<<<grid, blk, 0, stream>>>(
            deter, D, DG, (long long)DG,            // A0: deter slice, K0 = 512
            xpre, H3, 0,                             // A1: shared x, K 512..3583
            dyn_W, DG, (long long)ING * DG,          // B per-group stride
            ypre, D, (long long)DG,                  // C column block per group
            ING);
    }

    // ---- LN + SiLU over full D (in place) ----
    ln_silu_kernel<<<B, blk, 0, stream>>>(ypre, D, dyn_b, g_dyn, be_dyn, D);

    // ---- GRU gate projection: per group, y[:,g*DG:] @ gru_W[g] -> z cols ----
    {
        dim3 grid((3 * DG) / BN, B / BM, G);
        wmma_gemm_split<<<grid, blk, 0, stream>>>(
            ypre, D, DG, (long long)DG,              // A0 only (K0 == K)
            ypre, D, 0,
            gru_W, 3 * DG, (long long)DG * 3 * DG,
            z, D3, (long long)(3 * DG),
            DG);
    }

    // ---- gates + blend ----
    {
        int total = B * D;
        gru_gate_kernel<<<total / 256, blk, 0, stream>>>(z, gru_b, deter, out);
    }
    (void)in_sizes; (void)n_in; (void)out_size; (void)ws_size;
}
